// VectorQuantizer_36644660969912
// MI455X (gfx1250) — matrix operand order repaired and checked by reading, not compiled, not run on hardware
//
#include <hip/hip_runtime.h>
#include <cfloat>

// ---------------------------------------------------------------------------
// VQ-VAE vector quantizer for MI455X (gfx1250, wave32, WMMA).
//
// B = 32768 rows (z_e), K = 8192 codes, D = 256 dims.
// Dominant cost: 32768x8192x256 GEMM (137 GFLOP) -> compute bound
// (72 MB of HBM traffic is ~3us at 23.3 TB/s). We run the GEMM on the
// bf16 WMMA pipe (v_wmma_f32_16x16x32_bf16, 16384 FLOP/instr) with an
// error-compensated hi/lo split of the fp32 data:
//     cross ~= hi_z*hi_e + hi_z*lo_e + lo_z*hi_e   (~2^-16 rel. error)
// The three terms run in THREE INDEPENDENT accumulator chains (interleaved
// per k-step) so dependent WMMAs are 3 instructions apart -> no serial
// D->C stall on the XDL pipe. ||z||^2 is row-constant and dropped from the
// argmin metric; e_sq is computed exactly in fp32.
// ---------------------------------------------------------------------------

#define BQ   32768   // number of z_e rows
#define KC   8192    // number of codebook entries
#define DD   256     // embedding dim

typedef __attribute__((ext_vector_type(16))) __bf16 v16bf;
typedef __attribute__((ext_vector_type(8)))  __bf16 bf16x8;
typedef __attribute__((ext_vector_type(8)))  float  v8f;

// -------------------------- block reduction helper -------------------------
__device__ inline float block_reduce_sum_256(float v, float* sm) {
    const int lane = threadIdx.x & 31;
    const int wv   = threadIdx.x >> 5;
    #pragma unroll
    for (int o = 16; o >= 1; o >>= 1) v += __shfl_xor(v, o, 32);
    if (lane == 0) sm[wv] = v;
    __syncthreads();
    if (wv == 0) {
        v = (lane < 8) ? sm[lane] : 0.0f;
        #pragma unroll
        for (int o = 4; o >= 1; o >>= 1) v += __shfl_xor(v, o, 32);
    }
    return v;   // valid in thread 0
}

// ------------------- 1) split codebook to bf16 hi/lo + e_sq ----------------
__global__ void vq_prep_codebook(const float* __restrict__ cb,
                                 __bf16* __restrict__ e_hi,
                                 __bf16* __restrict__ e_lo,
                                 float*  __restrict__ e_sq) {
    __shared__ float sm[8];
    const int k = blockIdx.x;
    const int t = threadIdx.x;
    const size_t off = (size_t)k * DD + t;
    const float x  = cb[off];
    const __bf16 h = (__bf16)x;
    e_hi[off] = h;
    e_lo[off] = (__bf16)(x - (float)h);
    const float s = block_reduce_sum_256(x * x, sm);
    if (t == 0) e_sq[k] = s;
}

// -------------------------- 2) GEMM + fused argmin -------------------------
__device__ inline v16bf load_bfrag16(const __bf16* __restrict__ p) {
    bf16x8 p0 = *(const bf16x8*)(p);
    bf16x8 p1 = *(const bf16x8*)(p + 16);
    v16bf r;
    #pragma unroll
    for (int i = 0; i < 8; ++i) { r[i] = p0[i]; r[8 + i] = p1[i]; }
    return r;
}

__global__ void __launch_bounds__(256)
vq_gemm_argmin(const float* __restrict__ z_e,
               const __bf16* __restrict__ e_hi,
               const __bf16* __restrict__ e_lo,
               const float*  __restrict__ e_sq,
               int*   __restrict__ idx_out,
               float* __restrict__ idx_f_out) {
    const int wave   = threadIdx.x >> 5;          // 0..7
    const int lane   = threadIdx.x & 31;          // 0..31
    const int nlane  = lane & 15;                 // column-within-tile (C: N=lane%16)
    const int half8  = (lane < 16) ? 0 : 8;       // A k-half offset / C m offset
    const int ksel   = half8;                     // 0 or 8 (B k-half offset)
    const int rowbase = blockIdx.x * 128 + wave * 16;

    // ---- build A fragments for this wave's 16 rows x 256 dims, hi+lo split.
    // ISA 16-bit A 16x32 layout: elem e<8 -> K = kb + half8 + e
    //                            elem e>=8 -> K = kb + 16 + half8 + (e-8)
    v16bf a_hi[8], a_lo[8];
    {
        const float* __restrict__ zrow = z_e + (size_t)(rowbase + nlane) * DD;
        #pragma unroll
        for (int ks = 0; ks < 8; ++ks) {
            const int kb = ks * 32;
            #pragma unroll
            for (int half = 0; half < 2; ++half) {
                const int k0 = kb + half * 16 + half8;
                #pragma unroll
                for (int i = 0; i < 8; i += 2) {
                    const float2 p = *(const float2*)(zrow + k0 + i);
                    const __bf16 h0 = (__bf16)p.x;
                    const __bf16 h1 = (__bf16)p.y;
                    a_hi[ks][half * 8 + i]     = h0;
                    a_hi[ks][half * 8 + i + 1] = h1;
                    a_lo[ks][half * 8 + i]     = (__bf16)(p.x - (float)h0);
                    a_lo[ks][half * 8 + i + 1] = (__bf16)(p.y - (float)h1);
                }
            }
        }
    }

    float best[8];
    int   bidx[8];
    #pragma unroll
    for (int r = 0; r < 8; ++r) { best[r] = FLT_MAX; bidx[r] = 0; }

    // ---- sweep all 8192 codes in 16-wide N tiles
    for (int nt = 0; nt < KC / 16; ++nt) {
        const int c = nt * 16 + nlane;                  // this lane's code column
        const __bf16* __restrict__ bh_ptr = e_hi + (size_t)c * DD + ksel;
        const __bf16* __restrict__ bl_ptr = e_lo + (size_t)c * DD + ksel;
        // hint next tile's codebook rows into cache (global_prefetch_b8)
        __builtin_prefetch(bh_ptr + (size_t)16 * DD, 0, 1);

        // three independent accumulator chains: hi*hi, hi*lo, lo*hi
        v8f acc_hh, acc_hl, acc_lh;
        #pragma unroll
        for (int r = 0; r < 8; ++r) { acc_hh[r] = 0.0f; acc_hl[r] = 0.0f; acc_lh[r] = 0.0f; }

        #pragma unroll
        for (int ks = 0; ks < 8; ++ks) {
            const int off = ks * 32;
            const v16bf bh = load_bfrag16(bh_ptr + off);
            const v16bf bl = load_bfrag16(bl_ptr + off);
            acc_hh = __builtin_amdgcn_wmma_f32_16x16x32_bf16(
                         false, a_hi[ks], false, bh, (short)0, acc_hh, false, false);
            acc_hl = __builtin_amdgcn_wmma_f32_16x16x32_bf16(
                         false, a_hi[ks], false, bl, (short)0, acc_hl, false, false);
            acc_lh = __builtin_amdgcn_wmma_f32_16x16x32_bf16(
                         false, a_lo[ks], false, bh, (short)0, acc_lh, false, false);
        }

        // d2' = e_sq[c] - 2*cross   (||z||^2 dropped: row-constant for argmin)
        const float esq = e_sq[c];
        #pragma unroll
        for (int r = 0; r < 8; ++r) {
            const float cross = acc_hh[r] + acc_hl[r] + acc_lh[r];
            const float d = fmaf(-2.0f, cross, esq);
            const bool better = d < best[r];
            bidx[r] = better ? c : bidx[r];
            best[r] = better ? d : best[r];
        }
    }

    // ---- cross-lane argmin over the 16 columns sharing each row group.
    // C layout: lanes 0-15 hold M=r, lanes 16-31 hold M=8+r (same halves as half8).
    #pragma unroll
    for (int o = 8; o >= 1; o >>= 1) {
        #pragma unroll
        for (int r = 0; r < 8; ++r) {
            const float ov = __shfl_xor(best[r], o, 32);
            const int   oi = __shfl_xor(bidx[r], o, 32);
            if (ov < best[r] || (ov == best[r] && oi < bidx[r])) {
                best[r] = ov; bidx[r] = oi;
            }
        }
    }
    if (nlane == 0) {
        #pragma unroll
        for (int r = 0; r < 8; ++r) {
            const int row = rowbase + half8 + r;
            idx_out[row]   = bidx[r];
            idx_f_out[row] = (float)bidx[r];
        }
    }
}

// ---------------- 3) gather z_q + per-row squared-error partials -----------
__global__ void vq_gather_loss(const float* __restrict__ z_e,
                               const float* __restrict__ cb,
                               const int*   __restrict__ idx,
                               float* __restrict__ out_zq,
                               float* __restrict__ parts) {
    __shared__ float sm[8];
    const int b = blockIdx.x;
    const int t = threadIdx.x;
    const int k = idx[b];
    const float zq = cb[(size_t)k * DD + t];
    const float ze = z_e[(size_t)b * DD + t];
    out_zq[(size_t)b * DD + t] = zq;     // straight-through value == z_q
    const float d = zq - ze;
    const float s = block_reduce_sum_256(d * d, sm);
    if (t == 0) parts[b] = s;
}

// ---------------- 4) deterministic final loss reduction --------------------
__global__ void vq_final_loss(const float* __restrict__ parts,
                              float* __restrict__ out_loss) {
    __shared__ float sm[8];
    float s = 0.0f;
    for (int i = threadIdx.x; i < BQ; i += 256) s += parts[i];
    s = block_reduce_sum_256(s, sm);
    if (threadIdx.x == 0) {
        // vq_loss = codebook_loss + BETA*commitment_loss = 1.25 * mean(sq)
        *out_loss = 1.25f * s / (float)((size_t)BQ * DD);
    }
}

// ---------------------------------------------------------------------------
extern "C" void kernel_launch(void* const* d_in, const int* in_sizes, int n_in,
                              void* d_out, int out_size, void* d_ws, size_t ws_size,
                              hipStream_t stream) {
    const float* z_e = (const float*)d_in[0];   // (32768, 256) fp32
    const float* cb  = (const float*)d_in[1];   // (8192, 256)  fp32
    float* out = (float*)d_out;

    // d_out layout (tuple flattened): z_q_st (B*D f32) | indices (B as f32) | vq_loss (1)
    float* out_zq    = out;
    float* out_idx_f = out + (size_t)BQ * DD;
    float* out_loss  = out + (size_t)BQ * DD + BQ;

    // workspace layout (~8.3 MB)
    char* ws = (char*)d_ws;
    __bf16* e_hi = (__bf16*)ws;                                   // 4 MB
    __bf16* e_lo = e_hi + (size_t)KC * DD;                        // 4 MB
    float*  e_sq = (float*)(e_lo + (size_t)KC * DD);              // 32 KB
    int*    idxw = (int*)(e_sq + KC);                             // 128 KB
    float*  part = (float*)(idxw + BQ);                           // 128 KB

    vq_prep_codebook<<<dim3(KC), dim3(256), 0, stream>>>(cb, e_hi, e_lo, e_sq);
    vq_gemm_argmin<<<dim3(BQ / 128), dim3(256), 0, stream>>>(
        z_e, e_hi, e_lo, e_sq, idxw, out_idx_f);
    vq_gather_loss<<<dim3(BQ), dim3(256), 0, stream>>>(z_e, cb, idxw, out_zq, part);
    vq_final_loss<<<dim3(1), dim3(256), 0, stream>>>(part, out_loss);
}
